// AdaptiveModulator_52158082843109
// MI455X (gfx1250) — 8-bit matrix operand order repaired and checked by reading, compiled, not run on hardware
//
#include <hip/hip_runtime.h>
#include <hip/hip_bf16.h>
#include <stdint.h>

typedef __attribute__((ext_vector_type(8))) int v8i;

#define GB   32768
#define GM   2048
#define QA   255.0f
#define QW   127.0f


__device__ __forceinline__ float waveMaxF(float v) {
#pragma unroll
    for (int off = 16; off > 0; off >>= 1)
        v = fmaxf(v, __shfl_xor(v, off, 32));
    return v;
}


__global__ __launch_bounds__(256) void absmax_k(const float* __restrict__ x, int n,
                                                unsigned* __restrict__ dst) {
    float m = 0.f;
    for (int i = blockIdx.x * blockDim.x + threadIdx.x; i < n; i += gridDim.x * blockDim.x)
        m = fmaxf(m, fabsf(x[i]));
    m = waveMaxF(m);
    if ((threadIdx.x & 31) == 0) atomicMax(dst, __float_as_uint(m));
}


__global__ __launch_bounds__(256) void quantw_k(const float* __restrict__ W, int n,
                                                const unsigned* __restrict__ scal, int sidx,
                                                int8_t* __restrict__ out) {
    float s   = __uint_as_float(scal[sidx]) * (1.f / QW);
    float inv = 1.f / s;
    for (int i = blockIdx.x * blockDim.x + threadIdx.x; i < n; i += gridDim.x * blockDim.x) {
        float q = rintf(W[i] * inv);
        q = fminf(fmaxf(q, -QW), QW);
        out[i] = (int8_t)q;
    }
}


__global__ __launch_bounds__(256) void layer1_k(const float* __restrict__ snr,
                                                const float* __restrict__ W1,
                                                const float* __restrict__ b1,
                                                const unsigned* __restrict__ scal,
                                                float* __restrict__ out,
                                                unsigned* __restrict__ amax) {
    float s0    = __uint_as_float(scal[0]) * (1.f / QA);
    float w1s   = __uint_as_float(scal[1]) * (1.f / QW);
    float s_out = s0 * w1s;
    float inv_o = 1.f / s_out;

    size_t i4 = ((size_t)blockIdx.x * blockDim.x + threadIdx.x) * 4;
    int b = (int)(i4 >> 11);
    int m = (int)(i4 & (GM - 1));

    float xint = rintf(snr[b] / s0);
    float4 w  = *reinterpret_cast<const float4*>(W1 + m);
    float4 bb = *reinterpret_cast<const float4*>(b1 + m);

    float lmax = 0.f;
    float4 y;
    {
        float wi, bi;
        wi = fminf(fmaxf(rintf(w.x / w1s), -QW), QW); bi = rintf(bb.x * inv_o);
        y.x = fmaxf((xint * wi + bi) * s_out, 0.f);
        wi = fminf(fmaxf(rintf(w.y / w1s), -QW), QW); bi = rintf(bb.y * inv_o);
        y.y = fmaxf((xint * wi + bi) * s_out, 0.f);
        wi = fminf(fmaxf(rintf(w.z / w1s), -QW), QW); bi = rintf(bb.z * inv_o);
        y.z = fmaxf((xint * wi + bi) * s_out, 0.f);
        wi = fminf(fmaxf(rintf(w.w / w1s), -QW), QW); bi = rintf(bb.w * inv_o);
        y.w = fmaxf((xint * wi + bi) * s_out, 0.f);
    }
    *reinterpret_cast<float4*>(out + i4) = y;
    lmax = fmaxf(fmaxf(y.x, y.y), fmaxf(y.z, y.w));
    lmax = waveMaxF(lmax);
    if ((threadIdx.x & 31) == 0) atomicMax(amax, __float_as_uint(lmax));
}


__global__ __launch_bounds__(256) void quanta_k(const float* __restrict__ y,
                                                const unsigned* __restrict__ scal, int sidx,
                                                uint8_t* __restrict__ out) {
    float s   = __uint_as_float(scal[sidx]) * (1.f / QA);
    float inv = 1.f / s;
    size_t i4 = ((size_t)blockIdx.x * blockDim.x + threadIdx.x) * 4;
    float4 v = *reinterpret_cast<const float4*>(y + i4);
    uchar4 q;
    q.x = (uint8_t)fminf(rintf(v.x * inv), QA);
    q.y = (uint8_t)fminf(rintf(v.y * inv), QA);
    q.z = (uint8_t)fminf(rintf(v.z * inv), QA);
    q.w = (uint8_t)fminf(rintf(v.w * inv), QA);
    *reinterpret_cast<uchar4*>(out + i4) = q;
}


__global__ __launch_bounds__(256) void qgemm_k(const uint8_t* __restrict__ A,
                                               const int8_t*  __restrict__ Wq,
                                               const float*   __restrict__ bias,
                                               const unsigned* __restrict__ scal,
                                               int aIdx, int wIdx,
                                               float* __restrict__ out,
                                               unsigned* __restrict__ amax,
                                               int relu) {
    const int K = GM, N = GM;
    const int lane = threadIdx.x & 31;
    const int wid  = blockIdx.x * (blockDim.x >> 5) + (threadIdx.x >> 5);
    const int mtile = wid >> 5;
    const int nt64  = wid & 31;
    const int mbase = mtile << 4;
    const int nbase = nt64 << 6;

    float s_act = __uint_as_float(scal[aIdx]) * (1.f / QA);
    float w_s   = __uint_as_float(scal[wIdx]) * (1.f / QW);
    float s_out = s_act * w_s;

    const uint8_t* aptr = A + (size_t)(mbase + (lane & 15)) * K + ((lane >> 4) << 4);
    const int bko = (lane >> 4) << 4;
    const int8_t* bptr[4];
#pragma unroll
    for (int t = 0; t < 4; ++t)
        bptr[t] = Wq + (size_t)(nbase + t * 16 + (lane & 15)) * K + bko;

    v8i acc[4] = {};

    for (int k0 = 0; k0 < K; k0 += 64) {
        int4 a0 = *reinterpret_cast<const int4*>(aptr + k0 + 0);
        int4 a1 = *reinterpret_cast<const int4*>(aptr + k0 + 32);
        v8i av;
        av[0] = a0.x; av[1] = a0.y; av[2] = a0.z; av[3] = a0.w;
        av[4] = a1.x; av[5] = a1.y; av[6] = a1.z; av[7] = a1.w;
#pragma unroll
        for (int t = 0; t < 4; ++t) {
            int4 b0 = *reinterpret_cast<const int4*>(bptr[t] + k0);
            int4 b1 = *reinterpret_cast<const int4*>(bptr[t] + k0 + 32);
            v8i bv;
            bv[0] = b0.x; bv[1] = b0.y; bv[2] = b0.z; bv[3] = b0.w;
            bv[4] = b1.x; bv[5] = b1.y; bv[6] = b1.z; bv[7] = b1.w;
            acc[t] = __builtin_amdgcn_wmma_i32_16x16x64_iu8(false, av, true, bv, acc[t],
                                                            false, false);
        }
    }

    float lmax = 0.f;
    const int rowoff = (lane >> 4) << 3;
#pragma unroll
    for (int t = 0; t < 4; ++t) {
        int col = nbase + t * 16 + (lane & 15);
        float bint = rintf(bias[col] / s_out);
#pragma unroll
        for (int r = 0; r < 8; ++r) {
            float y = ((float)acc[t][r] + bint) * s_out;
            if (relu) y = fmaxf(y, 0.f);
            lmax = fmaxf(lmax, fabsf(y));
            out[(size_t)(mbase + rowoff + r) * N + col] = y;
        }
    }
    lmax = waveMaxF(lmax);
    if (lane == 0) atomicMax(amax, __float_as_uint(lmax));
}


__global__ __launch_bounds__(256) void finsig_k(float* __restrict__ y,
                                                const unsigned* __restrict__ scal, int sidx,
                                                unsigned* __restrict__ amax) {
    float s3  = __uint_as_float(scal[sidx]) * (1.f / QA);
    float inv = 1.f / s3;
    size_t i4 = ((size_t)blockIdx.x * blockDim.x + threadIdx.x) * 4;
    float4 v = *reinterpret_cast<float4*>(y + i4);
    float* c = &v.x;
    float lmax = 0.f;
#pragma unroll
    for (int j = 0; j < 4; ++j) {
        float xq = fminf(fmaxf(rintf(c[j] * inv), -QA), QA) * s3;
        float sg = 1.f / (1.f + __expf(-xq));
        c[j] = rintf(sg * QA) * (1.f / QA);
        lmax = fmaxf(lmax, c[j]);
    }
    *reinterpret_cast<float4*>(y + i4) = v;
    lmax = waveMaxF(lmax);
    if ((threadIdx.x & 31) == 0) atomicMax(amax, __float_as_uint(lmax));
}

__global__ __launch_bounds__(256) void finreq_k(float* __restrict__ y,
                                                const unsigned* __restrict__ scal,
                                                float* __restrict__ scale_out) {
    float s4  = __uint_as_float(scal[7]) * (1.f / QA);
    float inv = 1.f / s4;
    size_t tid = (size_t)blockIdx.x * blockDim.x + threadIdx.x;
    size_t i4 = tid * 4;
    float4 v = *reinterpret_cast<float4*>(y + i4);
    v.x = fminf(rintf(v.x * inv), QA) * s4;
    v.y = fminf(rintf(v.y * inv), QA) * s4;
    v.z = fminf(rintf(v.z * inv), QA) * s4;
    v.w = fminf(rintf(v.w * inv), QA) * s4;
    *reinterpret_cast<float4*>(y + i4) = v;
    if (tid == 0) *scale_out = s4;
}


extern "C" void kernel_launch(void* const* d_in, const int* in_sizes, int n_in,
                              void* d_out, int out_size, void* d_ws, size_t ws_size,
                              hipStream_t stream) {
    const float* snr = (const float*)d_in[0];
    const float* W1  = (const float*)d_in[1];
    const float* b1  = (const float*)d_in[2];
    const float* W2  = (const float*)d_in[3];
    const float* b2  = (const float*)d_in[4];
    const float* W3  = (const float*)d_in[5];
    const float* b3  = (const float*)d_in[6];

    float* outf = (float*)d_out;
    const size_t NTOT = (size_t)GB * GM;

    unsigned* scal = (unsigned*)d_ws;
    int8_t*  w2q  = (int8_t*)d_ws + 256;
    int8_t*  w3q  = w2q + (size_t)GM * GM;
    uint8_t* actq = (uint8_t*)(w3q + (size_t)GM * GM);

    hipMemsetAsync(d_ws, 0, 256, stream);

    const int nW = GM * GM;
    absmax_k<<<128, 256, 0, stream>>>(snr, GB, scal + 0);
    absmax_k<<<8,   256, 0, stream>>>(W1,  GM, scal + 1);
    absmax_k<<<4096, 256, 0, stream>>>(W2, nW, scal + 2);
    absmax_k<<<4096, 256, 0, stream>>>(W3, nW, scal + 3);
    quantw_k<<<4096, 256, 0, stream>>>(W2, nW, scal, 2, w2q);
    quantw_k<<<4096, 256, 0, stream>>>(W3, nW, scal, 3, w3q);

    const int elBlocks = (int)(NTOT / 4 / 256);
    layer1_k<<<elBlocks, 256, 0, stream>>>(snr, W1, b1, scal, outf, scal + 4);
    quanta_k<<<elBlocks, 256, 0, stream>>>(outf, scal, 4, actq);
    qgemm_k<<<8192, 256, 0, stream>>>(actq, w2q, b2, scal, 4, 2, outf, scal + 5, 1);
    quanta_k<<<elBlocks, 256, 0, stream>>>(outf, scal, 5, actq);
    qgemm_k<<<8192, 256, 0, stream>>>(actq, w3q, b3, scal, 5, 3, outf, scal + 6, 0);
    finsig_k<<<elBlocks, 256, 0, stream>>>(outf, scal, 6, scal + 7);
    finreq_k<<<elBlocks, 256, 0, stream>>>(outf, scal, outf + NTOT);
}
